// Step31MultiViewObservationEncoder_24833500906084
// MI455X (gfx1250) — matrix operand order repaired and checked by reading, not compiled, not run on hardware
//
#include <hip/hip_runtime.h>

typedef __attribute__((ext_vector_type(16))) _Float16 v16h;
typedef __attribute__((ext_vector_type(8)))  float    v8f;
typedef __attribute__((ext_vector_type(4)))  int      v4i;

#define B_  4
#define V_  3
#define N_  256
#define D_  32
#define H_  128
#define NEG_INF_ (-1.0e9f)
#define EPS_ 1e-4f

// CDNA5 async global->LDS path (probe via __has_builtin; fallback keeps compile green)
#if defined(__has_builtin)
#if __has_builtin(__builtin_amdgcn_global_load_async_to_lds_b128) && \
    __has_builtin(__builtin_amdgcn_s_wait_asynccnt)
#define USE_ASYNC_LDS 1
#endif
#endif

#ifdef USE_ASYNC_LDS
typedef __attribute__((address_space(1))) v4i gv4i;   // global (AS1)
typedef __attribute__((address_space(3))) v4i lv4i;   // LDS (AS3)
static __device__ __forceinline__ void async_g2l_b128(const void* g, void* l) {
  // global_load_async_to_lds_b128: (global src, lds dst, imm offset, cpol)
  __builtin_amdgcn_global_load_async_to_lds_b128(
      (gv4i*)(uintptr_t)g, (lv4i*)(uintptr_t)l, 0, 0);
}
#endif

static __device__ __forceinline__ v8f wmma32(v16h a, v16h b, v8f c) {
  // D = A(16x32 f16) * B(32x16 f16) + C(16x16 f32)
  return __builtin_amdgcn_wmma_f32_16x16x32_f16(false, a, false, b, (short)0, c, false, false);
}

// ---------------------------------------------------------------------------
// f32 -> f16 conversion
__global__ void f32_to_f16_k(const float* __restrict__ src, _Float16* __restrict__ dst, int n) {
  int i = blockIdx.x * 256 + threadIdx.x;
  if (i < n) dst[i] = (_Float16)src[i];
}

// ---------------------------------------------------------------------------
// Pack weight W[K,128] row-major f32 into WMMA B-fragment layout f16.
// P[((c*8 + t)*32 + lane)*16 + e] = W[c*32 + (lane/16)*8 + (e%8) + (e/8)*16, t*16 + lane%16]
// so each lane loads its v16h B-fragment as one contiguous 32-byte read.
__global__ void pack_w_k(const float* __restrict__ W, _Float16* __restrict__ P, int K, int KC) {
  int idx = blockIdx.x * 256 + threadIdx.x;
  int tot = KC * 4096;               // KC * 8 ntiles * 32 lanes * 16 elems
  if (idx >= tot) return;
  int e = idx & 15;
  int l = (idx >> 4) & 31;
  int t = (idx >> 9) & 7;
  int c = idx >> 12;
  int k = c * 32 + (l >> 4) * 8 + (e & 7) + (e >> 3) * 16;
  int n = t * 16 + (l & 15);
  P[idx] = (_Float16)((k < K) ? W[k * 128 + n] : 0.f);
}

// ---------------------------------------------------------------------------
// Generic WMMA GEMM: C[M,128] = act(A[M,Kpad] @ W + bias)
// One wave computes 16 rows x all 128 cols (8 accumulators). Loop-carried
// pointers + constant instruction offsets keep the inner loop free of
// per-load address math.
__global__ void gemm_wmma_k(const _Float16* __restrict__ A, int lda,
                            const _Float16* __restrict__ Wp,
                            const float* __restrict__ bias,
                            int KC, int M,
                            float* __restrict__ Cf, int ldc,
                            _Float16* __restrict__ Ch, int ldh, int coloff,
                            int relu)
{
  int lane = threadIdx.x & 31;
  int mt = (blockIdx.x * blockDim.x + threadIdx.x) >> 5;   // wave id == m-tile
  if (mt * 16 >= M) return;
  int half_ = lane >> 4;
  int n16   = lane & 15;
  int row   = mt * 16 + n16;         // A-fragment row for this lane
  v8f zero8 = {0.f, 0.f, 0.f, 0.f, 0.f, 0.f, 0.f, 0.f};
  v8f acc[8];
#pragma unroll
  for (int t = 0; t < 8; ++t) acc[t] = zero8;

  const _Float16* ap  = A + (size_t)row * lda + half_ * 8;
  const _Float16* wpp = Wp + (size_t)lane * 16;
  for (int c = 0; c < KC; ++c) {
    union { v16h v; uint4 q[2]; } a;
    a.q[0] = *(const uint4*)(ap);          // k = c*32 + half*8 + {0..7}
    a.q[1] = *(const uint4*)(ap + 16);     // k = c*32 + 16 + half*8 + {0..7}
#pragma unroll
    for (int t = 0; t < 8; ++t) {
      union { v16h v; uint4 q[2]; } b;
      const uint4* wp = (const uint4*)(wpp + (size_t)t * 512);
      b.q[0] = wp[0];
      b.q[1] = wp[1];
      acc[t] = wmma32(a.v, b.v, acc[t]);
    }
    ap  += 32;
    wpp += 4096;
  }
#pragma unroll
  for (int t = 0; t < 8; ++t) {
    int col = t * 16 + n16;
    float bs = bias ? bias[col] : 0.f;
#pragma unroll
    for (int v = 0; v < 8; ++v) {
      int r = mt * 16 + v + 8 * half_;
      float val = acc[t][v] + bs;
      if (relu) val = fmaxf(val, 0.f);
      if (Cf) Cf[(size_t)r * ldc + col] = val;
      if (Ch) Ch[(size_t)r * ldh + coloff + col] = (_Float16)val;
    }
  }
}

// ---------------------------------------------------------------------------
// Batched message passing: m[bv] = A[bv] @ h[bv], A = rel (256x256 f16/graph),
// h read from hcat cols 0..127 (lda 256), m written to hcat cols 128..255.
// One wave = 16 rows x 128 cols; B-gather uses one base pointer per chunk
// with immediate offsets (e*512 + t*32 bytes).
__global__ void mp_gemm_k(const _Float16* __restrict__ rel16,
                          const _Float16* __restrict__ hbuf,
                          _Float16* __restrict__ outbuf)
{
  int lane = threadIdx.x & 31;
  int mt = (blockIdx.x * blockDim.x + threadIdx.x) >> 5;
  if (mt * 16 >= 12 * N_) return;
  int half_ = lane >> 4;
  int n16   = lane & 15;
  int row = mt * 16 + n16;
  int bv  = row >> 8;
  int ar  = row & 255;
  v8f zero8 = {0.f, 0.f, 0.f, 0.f, 0.f, 0.f, 0.f, 0.f};
  v8f acc[8];
#pragma unroll
  for (int t = 0; t < 8; ++t) acc[t] = zero8;

  const _Float16* ap = rel16 + ((size_t)(bv * 256 + ar)) * 256 + half_ * 16;
  const _Float16* hb = hbuf + ((size_t)(bv * 256 + half_ * 16)) * 256 + n16;
  for (int c = 0; c < 8; ++c) {       // K = 256
    union { v16h v; uint4 q[2]; } a;
    a.q[0] = *(const uint4*)(ap);
    a.q[1] = *(const uint4*)(ap + 8);
#pragma unroll
    for (int t = 0; t < 8; ++t) {
      v16h bfrag;
#pragma unroll
      for (int e = 0; e < 16; ++e)
        bfrag[e] = hb[(size_t)e * 256 + t * 16];
      acc[t] = wmma32(a.v, bfrag, acc[t]);
    }
    ap += 32;
    hb += 32 * 256;
  }
#pragma unroll
  for (int t = 0; t < 8; ++t) {
    int col = t * 16 + n16;
#pragma unroll
    for (int v = 0; v < 8; ++v) {
      int r = mt * 16 + v + 8 * half_;
      outbuf[(size_t)r * 256 + 128 + col] = (_Float16)acc[t][v];
    }
  }
}

// ---------------------------------------------------------------------------
// Latent mean/std over V=3 -> fused f16 [B*N, 256] (mean | std)
__global__ void latent_stats_k(const float* __restrict__ vl, _Float16* __restrict__ fused) {
  int idx = blockIdx.x * 256 + threadIdx.x;
  if (idx >= B_ * N_ * H_) return;
  int h = idx & 127;
  int n = (idx >> 7) & 255;
  int b = idx >> 15;
  size_t base = (((size_t)b * V_) * N_ + n) * H_ + h;
  const size_t vs = (size_t)N_ * H_;
  float x0 = vl[base], x1 = vl[base + vs], x2 = vl[base + 2 * vs];
  float mean = (x0 + x1 + x2) * (1.f / 3.f);
  float m2   = (x0 * x0 + x1 * x1 + x2 * x2) * (1.f / 3.f);
  float sd   = sqrtf(fmaxf(m2 - mean * mean, 0.f));
  size_t fb = ((size_t)(b * N_ + n)) * 256;
  fused[fb + h]       = (_Float16)mean;
  fused[fb + 128 + h] = (_Float16)sd;
}

// ---------------------------------------------------------------------------
// Evidence statistics over V=3 + logit hint + masked stat outputs
__global__ void ev_stats_k(const float* __restrict__ rel, const float* __restrict__ sup,
                           const float* __restrict__ wit, const float* __restrict__ bun,
                           _Float16* __restrict__ ev16, float* __restrict__ hint,
                           float* __restrict__ o_rm, float* __restrict__ o_rs,
                           float* __restrict__ o_sm, float* __restrict__ o_ss,
                           float* __restrict__ o_wm, float* __restrict__ o_ws)
{
  int idx = blockIdx.x * 256 + threadIdx.x;
  if (idx >= B_ * N_ * N_) return;
  int b  = idx >> 16;
  int ij = idx & 65535;
  int i = ij >> 8, j = ij & 255;
  size_t base = ((size_t)b * V_) * 65536 + ij;
  const size_t vs = 65536;
  const float inv3 = 1.f / 3.f;
  _Float16* ev = ev16 + (size_t)idx * 18;

  float r0 = rel[base], r1 = rel[base + vs], r2 = rel[base + 2 * vs];
  float rm = (r0 + r1 + r2) * inv3;
  float rs = sqrtf(fmaxf((r0 * r0 + r1 * r1 + r2 * r2) * inv3 - rm * rm, 0.f));
  float rmin = fminf(r0, fminf(r1, r2));
  float rmax = fmaxf(r0, fmaxf(r1, r2));

  float s0 = sup[base], s1 = sup[base + vs], s2 = sup[base + 2 * vs];
  float sm = (s0 + s1 + s2) * inv3;
  float ss = sqrtf(fmaxf((s0 * s0 + s1 * s1 + s2 * s2) * inv3 - sm * sm, 0.f));
  float smin = fminf(s0, fminf(s1, s2));
  float smax = fmaxf(s0, fmaxf(s1, s2));

  float w0 = wit[base], w1 = wit[base + vs], w2 = wit[base + 2 * vs];
  float wm = (w0 + w1 + w2) * inv3;
  float wsd = sqrtf(fmaxf((w0 * w0 + w1 * w1 + w2 * w2) * inv3 - wm * wm, 0.f));

  ev[0] = (_Float16)rm;  ev[1] = (_Float16)rs;  ev[2] = (_Float16)rmin; ev[3] = (_Float16)rmax;
  ev[4] = (_Float16)sm;  ev[5] = (_Float16)ss;  ev[6] = (_Float16)smin; ev[7] = (_Float16)smax;
  ev[8] = (_Float16)wm;  ev[9] = (_Float16)wsd;

#pragma unroll
  for (int ch = 0; ch < 4; ++ch) {
    size_t bb = base * 4 + ch;
    float b0 = bun[bb], b1v = bun[bb + vs * 4], b2 = bun[bb + 2 * vs * 4];
    float bm = (b0 + b1v + b2) * inv3;
    float bsd = sqrtf(fmaxf((b0 * b0 + b1v * b1v + b2 * b2) * inv3 - bm * bm, 0.f));
    ev[10 + ch] = (_Float16)bm;
    ev[14 + ch] = (_Float16)bsd;
  }

  float p = 0.5f * (rm + sm);
  p = fminf(fmaxf(p, EPS_), 1.f - EPS_);
  hint[idx] = logf(p) - log1pf(-p);

  bool diag = (i == j);
  o_rm[idx] = diag ? 0.f : rm;
  o_rs[idx] = diag ? 0.f : rs;
  o_sm[idx] = diag ? 0.f : sm;
  o_ss[idx] = diag ? 0.f : ss;
  o_wm[idx] = diag ? 0.f : wm;
  o_ws[idx] = diag ? 0.f : wsd;
}

// ---------------------------------------------------------------------------
// Edge head (dominant GEMM): per wave, 16 pairs (fixed i, j0..j0+15).
// Node rows staged to LDS via async global->LDS DMA.
// Features synthesized on the fly into WMMA A-fragments; K padded 530->544.
__global__ void edge_wmma_k(const _Float16* __restrict__ node16,
                            const _Float16* __restrict__ ev16,
                            const _Float16* __restrict__ Wp,
                            const float* __restrict__ b0,
                            const float* __restrict__ w1,
                            const float* __restrict__ b1,
                            const float* __restrict__ hint,
                            float* __restrict__ e_raw)
{
  __shared__ __align__(16) _Float16 sh[8][2176];   // per wave: hi(128) + 16x hj(128)
  int lane = threadIdx.x & 31;
  int wv   = threadIdx.x >> 5;
  int gw   = blockIdx.x * 8 + wv;
  int jt = gw & 15;
  int i  = (gw >> 4) & 255;
  int b  = gw >> 12;
  int j0 = jt << 4;

  uint4* dst = (uint4*)sh[wv];
  const uint4* hi_src = (const uint4*)(node16 + ((size_t)(b * N_ + i)) * H_);
  const uint4* hj_src = (const uint4*)(node16 + ((size_t)(b * N_ + j0)) * H_);
#ifdef USE_ASYNC_LDS
  if (lane < 16) async_g2l_b128(hi_src + lane, dst + lane);
#pragma unroll
  for (int t = 0; t < 8; ++t)
    async_g2l_b128(hj_src + t * 32 + lane, dst + 16 + t * 32 + lane);
  __builtin_amdgcn_s_wait_asynccnt(0);
#else
  if (lane < 16) dst[lane] = hi_src[lane];
#pragma unroll
  for (int t = 0; t < 8; ++t) dst[16 + t * 32 + lane] = hj_src[t * 32 + lane];
#endif
  __syncthreads();

  int half_ = lane >> 4;
  int m     = lane & 15;
  int j = j0 + m;
  const _Float16* hi = sh[wv];
  const _Float16* hj = sh[wv] + 128 + m * 128;
  size_t pair = ((size_t)(b * N_ + i)) * N_ + j;
  const _Float16* evp = ev16 + pair * 18;

  v8f zero8 = {0.f, 0.f, 0.f, 0.f, 0.f, 0.f, 0.f, 0.f};
  v8f acc[8];
#pragma unroll
  for (int t = 0; t < 8; ++t) acc[t] = zero8;

  const _Float16* wpp = Wp + (size_t)lane * 16;
  for (int c = 0; c < 17; ++c) {       // Kpad = 544
    v16h a;
#pragma unroll
    for (int e = 0; e < 16; ++e) {
      int k = c * 32 + ((e < 8) ? (half_ * 8 + e) : (16 + half_ * 8 + (e - 8)));
      float f;
      if (k < 128)      f = (float)hi[k];
      else if (k < 256) f = (float)hj[k - 128];
      else if (k < 384) { float x = (float)hi[k - 256], y = (float)hj[k - 256]; f = fabsf(x - y); }
      else if (k < 512) { float x = (float)hi[k - 384], y = (float)hj[k - 384]; f = x * y; }
      else if (k < 530) f = (float)evp[k - 512];
      else              f = 0.f;
      a[e] = (_Float16)f;
    }
    if (c + 1 < 17)
      __builtin_prefetch((const void*)(wpp + 4096), 0, 0);
#pragma unroll
    for (int t = 0; t < 8; ++t) {
      union { v16h v; uint4 q[2]; } bb;
      const uint4* wp = (const uint4*)(wpp + (size_t)t * 512);
      bb.q[0] = wp[0];
      bb.q[1] = wp[1];
      acc[t] = wmma32(a, bb.v, acc[t]);
    }
    wpp += 4096;
  }

  float eacc[8];
#pragma unroll
  for (int v = 0; v < 8; ++v) eacc[v] = 0.f;
#pragma unroll
  for (int t = 0; t < 8; ++t) {
    int col = t * 16 + m;
    float bias0 = b0[col];
    float wc = w1[col];
#pragma unroll
    for (int v = 0; v < 8; ++v) {
      float h = acc[t][v] + bias0;
      h = fmaxf(h, 0.f);
      eacc[v] += h * wc;
    }
  }
#pragma unroll
  for (int v = 0; v < 8; ++v) {
    float s = eacc[v];
    s += __shfl_xor(s, 1);
    s += __shfl_xor(s, 2);
    s += __shfl_xor(s, 4);
    s += __shfl_xor(s, 8);
    if (m == 0) {
      int jj = j0 + v + 8 * half_;
      size_t oidx = ((size_t)(b * N_ + i)) * N_ + jj;
      e_raw[oidx] = s + b1[0] + hint[oidx];
    }
  }
}

// ---------------------------------------------------------------------------
// Tiny final projections: type (3) and state (4)
__global__ void heads_final_k(const _Float16* __restrict__ t0, const _Float16* __restrict__ s0,
                              const float* __restrict__ th1_w, const float* __restrict__ th1_b,
                              const float* __restrict__ sh1_w, const float* __restrict__ sh1_b,
                              float* __restrict__ o_type, float* __restrict__ o_state)
{
  int idx = blockIdx.x * 256 + threadIdx.x;
  if (idx >= 1024 * 7) return;
  int nidx = idx / 7;
  int c = idx % 7;
  if (c < 3) {
    float s = th1_b[c];
    for (int k = 0; k < 128; ++k) s += (float)t0[(size_t)nidx * 128 + k] * th1_w[k * 3 + c];
    o_type[nidx * 3 + c] = s;
  } else {
    int cc = c - 3;
    float s = sh1_b[cc];
    for (int k = 0; k < 128; ++k) s += (float)s0[(size_t)nidx * 128 + k] * sh1_w[k * 4 + cc];
    o_state[nidx * 4 + cc] = s;
  }
}

// ---------------------------------------------------------------------------
// Symmetrize + diagonal mask
__global__ void sym_k(const float* __restrict__ e_raw, float* __restrict__ e_out) {
  int idx = blockIdx.x * 256 + threadIdx.x;
  if (idx >= B_ * N_ * N_) return;
  int b = idx >> 16;
  int i = (idx >> 8) & 255;
  int j = idx & 255;
  if (i == j) { e_out[idx] = NEG_INF_; return; }
  size_t tidx = ((size_t)b << 16) + ((size_t)j << 8) + i;
  e_out[idx] = 0.5f * (e_raw[idx] + e_raw[tidx]);
}

// ---------------------------------------------------------------------------
extern "C" void kernel_launch(void* const* d_in, const int* in_sizes, int n_in,
                              void* d_out, int out_size, void* d_ws, size_t ws_size,
                              hipStream_t stream) {
  const float* slots  = (const float*)d_in[0];
  const float* rel    = (const float*)d_in[1];
  const float* sup    = (const float*)d_in[2];
  const float* wit    = (const float*)d_in[3];
  const float* bundle = (const float*)d_in[4];
  const float* ve_in_w = (const float*)d_in[5];
  const float* ve_in_b = (const float*)d_in[6];
  const float* ve_l0_w = (const float*)d_in[7];
  const float* ve_l0_b = (const float*)d_in[8];
  const float* ve_l1_w = (const float*)d_in[9];
  const float* ve_l1_b = (const float*)d_in[10];
  const float* ve_l2_w = (const float*)d_in[11];
  const float* ve_l2_b = (const float*)d_in[12];
  const float* nf0_w = (const float*)d_in[13];
  const float* nf0_b = (const float*)d_in[14];
  const float* nf1_w = (const float*)d_in[15];
  const float* nf1_b = (const float*)d_in[16];
  const float* th0_w = (const float*)d_in[17];
  const float* th0_b = (const float*)d_in[18];
  const float* th1_w = (const float*)d_in[19];
  const float* th1_b = (const float*)d_in[20];
  const float* sh0_w = (const float*)d_in[21];
  const float* sh0_b = (const float*)d_in[22];
  const float* sh1_w = (const float*)d_in[23];
  const float* sh1_b = (const float*)d_in[24];
  const float* eh0_w = (const float*)d_in[25];
  const float* eh0_b = (const float*)d_in[26];
  const float* eh1_w = (const float*)d_in[27];
  const float* eh1_b = (const float*)d_in[28];

  float* out = (float*)d_out;
  float* o_node  = out;                 // [4,256,128]
  float* o_vl    = out + 131072;        // [4,3,256,128]
  float* o_type  = out + 524288;        // [4,256,3]
  float* o_state = out + 527360;        // [4,256,4]
  float* o_e     = out + 531456;        // [4,256,256]
  float* o_rm    = out + 793600;
  float* o_rs    = out + 1055744;
  float* o_sm    = out + 1317888;
  float* o_ss    = out + 1580032;
  float* o_wm    = out + 1842176;
  float* o_wsd   = out + 2104320;

  char* wsb = (char*)d_ws;
  size_t cur = 0;
  auto alloc = [&](size_t bytes) -> void* {
    void* p = wsb + cur;
    cur = (cur + bytes + 255) & ~(size_t)255;
    return p;
  };

  _Float16* rel16   = (_Float16*)alloc(786432 * 2);   // [12,256,256]
  _Float16* x16     = (_Float16*)alloc(98304 * 2);    // [3072,32]
  _Float16* hcatA   = (_Float16*)alloc(786432 * 2);   // [3072,256]
  _Float16* hcatB   = (_Float16*)alloc(786432 * 2);
  _Float16* fused16 = (_Float16*)alloc(262144 * 2);   // [1024,256]
  _Float16* nh016   = (_Float16*)alloc(131072 * 2);   // [1024,128]
  _Float16* node16  = (_Float16*)alloc(131072 * 2);
  _Float16* t016    = (_Float16*)alloc(131072 * 2);
  _Float16* s016    = (_Float16*)alloc(131072 * 2);
  _Float16* ev16    = (_Float16*)alloc(4718592 * 2);  // [4,256,256,18]
  float*    hint    = (float*)alloc(262144 * 4);
  float*    e_raw   = (float*)alloc(262144 * 4);
  _Float16* p_vein  = (_Float16*)alloc(4096 * 2);     // KC=1
  _Float16* p_l0    = (_Float16*)alloc(32768 * 2);    // KC=8
  _Float16* p_l1    = (_Float16*)alloc(32768 * 2);
  _Float16* p_l2    = (_Float16*)alloc(32768 * 2);
  _Float16* p_nf0   = (_Float16*)alloc(32768 * 2);
  _Float16* p_nf1   = (_Float16*)alloc(16384 * 2);    // KC=4
  _Float16* p_th0   = (_Float16*)alloc(16384 * 2);
  _Float16* p_sh0   = (_Float16*)alloc(16384 * 2);
  _Float16* p_eh0   = (_Float16*)alloc(69632 * 2);    // KC=17

  // Conversions
  f32_to_f16_k<<<(786432 + 255) / 256, 256, 0, stream>>>(rel, rel16, 786432);
  f32_to_f16_k<<<(98304 + 255) / 256, 256, 0, stream>>>(slots, x16, 98304);

  // Weight packing
  auto pack = [&](const float* W, _Float16* P, int K, int KC) {
    pack_w_k<<<(KC * 4096 + 255) / 256, 256, 0, stream>>>(W, P, K, KC);
  };
  pack(ve_in_w, p_vein, 32, 1);
  pack(ve_l0_w, p_l0, 256, 8);
  pack(ve_l1_w, p_l1, 256, 8);
  pack(ve_l2_w, p_l2, 256, 8);
  pack(nf0_w, p_nf0, 256, 8);
  pack(nf1_w, p_nf1, 128, 4);
  pack(th0_w, p_th0, 128, 4);
  pack(sh0_w, p_sh0, 128, 4);
  pack(eh0_w, p_eh0, 530, 17);

  // View encoder: in-proj then 3 message-passing layers
  // gemm grid: waves = M/16, 8 waves/block
  gemm_wmma_k<<<24, 256, 0, stream>>>(x16, 32, p_vein, ve_in_b, 1, 3072,
                                      nullptr, 0, hcatA, 256, 0, 1);
  mp_gemm_k<<<24, 256, 0, stream>>>(rel16, hcatA, hcatA);
  gemm_wmma_k<<<24, 256, 0, stream>>>(hcatA, 256, p_l0, ve_l0_b, 8, 3072,
                                      nullptr, 0, hcatB, 256, 0, 1);
  mp_gemm_k<<<24, 256, 0, stream>>>(rel16, hcatB, hcatB);
  gemm_wmma_k<<<24, 256, 0, stream>>>(hcatB, 256, p_l1, ve_l1_b, 8, 3072,
                                      nullptr, 0, hcatA, 256, 0, 1);
  mp_gemm_k<<<24, 256, 0, stream>>>(rel16, hcatA, hcatA);
  gemm_wmma_k<<<24, 256, 0, stream>>>(hcatA, 256, p_l2, ve_l2_b, 8, 3072,
                                      o_vl, 128, nullptr, 0, 0, 1);

  // Fusion + node
  latent_stats_k<<<512, 256, 0, stream>>>(o_vl, fused16);
  gemm_wmma_k<<<8, 256, 0, stream>>>(fused16, 256, p_nf0, nf0_b, 8, 1024,
                                     nullptr, 0, nh016, 128, 0, 1);
  gemm_wmma_k<<<8, 256, 0, stream>>>(nh016, 128, p_nf1, nf1_b, 4, 1024,
                                     o_node, 128, node16, 128, 0, 0);

  // Heads
  gemm_wmma_k<<<8, 256, 0, stream>>>(node16, 128, p_th0, th0_b, 4, 1024,
                                     nullptr, 0, t016, 128, 0, 1);
  gemm_wmma_k<<<8, 256, 0, stream>>>(node16, 128, p_sh0, sh0_b, 4, 1024,
                                     nullptr, 0, s016, 128, 0, 1);
  heads_final_k<<<28, 256, 0, stream>>>(t016, s016, th1_w, th1_b, sh1_w, sh1_b,
                                        o_type, o_state);

  // Evidence statistics + masked outputs + logit hint
  ev_stats_k<<<1024, 256, 0, stream>>>(rel, sup, wit, bundle, ev16, hint,
                                       o_rm, o_rs, o_sm, o_ss, o_wm, o_wsd);

  // Edge head (dominant WMMA GEMM), then symmetrize + diag mask
  edge_wmma_k<<<2048, 256, 0, stream>>>(node16, ev16, p_eh0, eh0_b, eh1_w, eh1_b,
                                        hint, e_raw);
  sym_k<<<1024, 256, 0, stream>>>(e_raw, o_e);
}
